// EdgeHTR_85323820302757
// MI455X (gfx1250) — matrix operand order repaired and checked by reading, not compiled, not run on hardware
//
#include <hip/hip_runtime.h>

#define D 128
typedef __bf16 bf16;
typedef __attribute__((ext_vector_type(16))) __bf16 v16bf;
typedef __attribute__((ext_vector_type(8)))  __bf16 v8bf;
typedef __attribute__((ext_vector_type(8)))  float  v8f;
typedef __attribute__((ext_vector_type(4)))  int    v4i;

// LDS row strides (elements), padded so row-stride % 64 dwords != 0 (bank spread)
#define XS_STRIDE 392   // 384 + 8
#define WS_STRIDE 136   // 128 + 8
#define HS_STRIDE 136

#if defined(__has_builtin)
#if __has_builtin(__builtin_amdgcn_global_load_async_to_lds_b128)
#define HAVE_ASYNC_LDS 1
#endif
#endif

union ABFrag { v16bf v; v8bf h[2]; };

// ---------------- prep kernels ----------------

__global__ void k_init_winner(int* __restrict__ w, int n) {
    int i = blockIdx.x * blockDim.x + threadIdx.x;
    if (i < n) w[i] = -1;
}

__global__ void k_winner(const int* __restrict__ e1map, int* __restrict__ w, int n) {
    int i = blockIdx.x * blockDim.x + threadIdx.x;
    if (i < n) atomicMax(&w[e1map[i]], i);   // last-write-wins == max edge id wins
}

__global__ void k_cvt_w(const float* __restrict__ W1, const float* __restrict__ W2,
                        bf16* __restrict__ w1bf, bf16* __restrict__ w2bf,
                        int n1, int n2) {
    int i = blockIdx.x * blockDim.x + threadIdx.x;
    if (i < n1)            w1bf[i]      = (bf16)W1[i];
    else if (i < n1 + n2)  w2bf[i - n1] = (bf16)W2[i - n1];
}

__global__ void k_copy4(const float4* __restrict__ src, float4* __restrict__ dst, long n4) {
    long i = (long)blockIdx.x * blockDim.x + threadIdx.x;
    long stride = (long)gridDim.x * blockDim.x;
    for (; i < n4; i += stride) dst[i] = src[i];
}

// ---------------- helpers ----------------

__device__ __forceinline__ void cvt_store16(bf16* dst, const float* src) {
    const float4* s4 = (const float4*)src;
#pragma unroll
    for (int i = 0; i < 2; ++i) {
        float4 f0 = s4[2 * i], f1 = s4[2 * i + 1];
        v8bf b;
        b[0] = (bf16)f0.x; b[1] = (bf16)f0.y; b[2] = (bf16)f0.z; b[3] = (bf16)f0.w;
        b[4] = (bf16)f1.x; b[5] = (bf16)f1.y; b[6] = (bf16)f1.z; b[7] = (bf16)f1.w;
        *(v8bf*)(dst + 8 * i) = b;
    }
}

__device__ __forceinline__ void zero_store16(bf16* dst) {
    v8bf z = {};
    *(v8bf*)(dst)     = z;
    *(v8bf*)(dst + 8) = z;
}

// ---------------- main fused MLP kernel ----------------

__launch_bounds__(256, 2)
__global__ void k_edge_mlp(const float* __restrict__ t_e2,
                           const float* __restrict__ h,
                           const int*   __restrict__ ei,      // [2, E1]
                           const int*   __restrict__ e1map,   // [E1]
                           const float* __restrict__ b1,
                           const float* __restrict__ b2,
                           const bf16*  __restrict__ w1g,     // [384,128] row-major bf16
                           const bf16*  __restrict__ w2g,     // [128,128] row-major bf16
                           const int*   __restrict__ winner,
                           float*       __restrict__ out,
                           int E1, int numTiles)
{
    extern __shared__ __align__(16) char smem[];
    bf16*  w1s = (bf16*)smem;                    // 384 x WS_STRIDE
    bf16*  xs  = w1s + 384 * WS_STRIDE;          //  32 x XS_STRIDE
    bf16*  hs  = xs  +  32 * XS_STRIDE;          //  32 x HS_STRIDE
    float* b1s = (float*)(hs + 32 * HS_STRIDE);  // 128
    float* b2s = b1s + 128;                      // 128
    int*   e2s = (int*)(b2s + 128);              // 32
    int*   wfs = e2s + 32;                       // 32

    const int tid = threadIdx.x;

    // Stage W1 (once per persistent block) with row padding.
#ifdef HAVE_ASYNC_LDS
    for (int c = tid; c < 384 * 128 / 8; c += 256) {
        int row = c >> 4, col = (c & 15) << 3;
        __builtin_amdgcn_global_load_async_to_lds_b128(
            (v4i*)(void*)(w1g + row * 128 + col),
            (v4i*)(void*)(w1s + row * WS_STRIDE + col), 0, 0);
    }
#if __has_builtin(__builtin_amdgcn_s_wait_asynccnt)
    __builtin_amdgcn_s_wait_asynccnt(0);
#else
    asm volatile("s_wait_asynccnt 0x0" ::: "memory");
#endif
#else
    for (int c = tid; c < 384 * 128 / 8; c += 256) {
        int row = c >> 4, col = (c & 15) << 3;
        *(v8bf*)(w1s + row * WS_STRIDE + col) = *(const v8bf*)(w1g + row * 128 + col);
    }
#endif
    if (tid < 128) { b1s[tid] = b1[tid]; b2s[tid] = b2[tid]; }

    const int lane  = tid & 31;
    const int wid   = tid >> 5;
    const int msub  = wid >> 2;           // 0..1  -> M sub-tile of 16
    const int nbase = (wid & 3) << 5;     // 0,32,64,96 -> 32 N cols per wave
    const int arow  = msub * 16 + (lane & 15);
    const int koff  = (lane >> 4) << 3;   // 0 or 8 (A-frag K-phase per ISA layout)

    const int gm = tid >> 3;              // gather row 0..31
    const int gc = (tid & 7) << 4;        // 16-float chunk within 128-col segment

    for (int tile = blockIdx.x; tile < numTiles; tile += gridDim.x) {
        __syncthreads();   // previous tile fully consumed (xs/hs/e2s free)

        // ---- gather 32 edges -> xs (bf16), plus per-row scatter metadata ----
        int eg = tile * 32 + gm;
        if ((tid & 7) == 0) {
            int e2 = (eg < E1) ? e1map[eg] : 0;
            e2s[gm] = e2;
            wfs[gm] = (eg < E1) && (winner[e2] == eg);
        }
        bf16* xrow = xs + gm * XS_STRIDE;
        if (eg < E1) {
            int  s  = ei[eg];
            int  d2 = ei[E1 + eg];
            long e2 = e1map[eg];
            cvt_store16(xrow + 0   + gc, h    + (long)s  * D + gc);
            cvt_store16(xrow + 128 + gc, h    + (long)d2 * D + gc);
            cvt_store16(xrow + 256 + gc, t_e2 + e2       * D + gc);
        } else {
            zero_store16(xrow + 0   + gc);
            zero_store16(xrow + 128 + gc);
            zero_store16(xrow + 256 + gc);
        }
        // ---- prefetch next tile's gather rows into L2/WGP$ ----
        {
            int en = eg + gridDim.x * 32;
            if (en < E1) {
                int  sn  = ei[en];
                int  dn  = ei[E1 + en];
                long e2n = e1map[en];
                __builtin_prefetch(h    + (long)sn * D + gc, 0, 1);
                __builtin_prefetch(h    + (long)dn * D + gc, 0, 1);
                __builtin_prefetch(t_e2 + e2n      * D + gc, 0, 1);
            }
        }
        __syncthreads();

        // ---- GEMM1: [32x384] x [384x128], bf16 -> f32 acc (W1 in LDS) ----
        v8f c0 = {}, c1 = {};
        for (int kk = 0; kk < 12; ++kk) {
            ABFrag a;
            const bf16* ap = xs + arow * XS_STRIDE + kk * 32;
            a.h[0] = *(const v8bf*)(ap + koff);
            a.h[1] = *(const v8bf*)(ap + 16 + koff);
            const bf16* bp = w1s + (kk * 32 + koff) * WS_STRIDE + nbase + (lane & 15);
            ABFrag b0, b1f;
#pragma unroll
            for (int s = 0; s < 8; ++s) {
                b0.v[s]      = bp[s * WS_STRIDE];
                b0.v[8 + s]  = bp[(16 + s) * WS_STRIDE];
                b1f.v[s]     = bp[s * WS_STRIDE + 16];
                b1f.v[8 + s] = bp[(16 + s) * WS_STRIDE + 16];
            }
            c0 = __builtin_amdgcn_wmma_f32_16x16x32_bf16(false, a.v, false, b0.v,  (short)0, c0, false, false);
            c1 = __builtin_amdgcn_wmma_f32_16x16x32_bf16(false, a.v, false, b1f.v, (short)0, c1, false, false);
        }

        // ---- epilogue 1: +bias, SiLU, write bf16 hmid tile ----
        {
            int   n0   = nbase + (lane & 15);
            int   mb   = msub * 16 + ((lane >> 4) << 3);
            float bia0 = b1s[n0], bia1 = b1s[n0 + 16];
#pragma unroll
            for (int r = 0; r < 8; ++r) {
                float v0 = c0[r] + bia0;
                float v1 = c1[r] + bia1;
                v0 = v0 / (1.f + __expf(-v0));
                v1 = v1 / (1.f + __expf(-v1));
                hs[(mb + r) * HS_STRIDE + n0]      = (bf16)v0;
                hs[(mb + r) * HS_STRIDE + n0 + 16] = (bf16)v1;
            }
        }
        __syncthreads();

        // ---- GEMM2: [32x128] x [128x128]; W2 B-frags from global (L2/WGP$-hot) ----
        v8f d0 = {}, d1 = {};
        for (int kk = 0; kk < 4; ++kk) {
            ABFrag a;
            const bf16* ap = hs + arow * HS_STRIDE + kk * 32;
            a.h[0] = *(const v8bf*)(ap + koff);
            a.h[1] = *(const v8bf*)(ap + 16 + koff);
            const bf16* bp = w2g + (kk * 32 + koff) * 128 + nbase + (lane & 15);
            ABFrag b0, b1f;
#pragma unroll
            for (int s = 0; s < 8; ++s) {
                b0.v[s]      = bp[s * 128];
                b0.v[8 + s]  = bp[(16 + s) * 128];
                b1f.v[s]     = bp[s * 128 + 16];
                b1f.v[8 + s] = bp[(16 + s) * 128 + 16];
            }
            d0 = __builtin_amdgcn_wmma_f32_16x16x32_bf16(false, a.v, false, b0.v,  (short)0, d0, false, false);
            d1 = __builtin_amdgcn_wmma_f32_16x16x32_bf16(false, a.v, false, b1f.v, (short)0, d1, false, false);
        }

        // ---- epilogue 2: t_new[e2] = sub_t + delta + b2 (winner only) ----
        {
            int   n0   = nbase + (lane & 15);
            int   mb   = msub * 16 + ((lane >> 4) << 3);
            float bia0 = b2s[n0], bia1 = b2s[n0 + 16];
#pragma unroll
            for (int r = 0; r < 8; ++r) {
                int mrow = mb + r;
                int e = tile * 32 + mrow;
                if (e < E1 && wfs[mrow]) {
                    long rr = (long)e2s[mrow] * D;
                    out[rr + n0]      = t_e2[rr + n0]      + d0[r] + bia0;
                    out[rr + n0 + 16] = t_e2[rr + n0 + 16] + d1[r] + bia1;
                }
            }
        }
    }
}

// ---------------- launcher ----------------

extern "C" void kernel_launch(void* const* d_in, const int* in_sizes, int n_in,
                              void* d_out, int out_size, void* d_ws, size_t ws_size,
                              hipStream_t stream) {
    const float* t_e2 = (const float*)d_in[0];
    const float* h    = (const float*)d_in[1];
    const int*   ei   = (const int*)d_in[2];
    const int*   e1m  = (const int*)d_in[3];
    const float* W1   = (const float*)d_in[4];
    const float* b1   = (const float*)d_in[5];
    const float* W2   = (const float*)d_in[6];
    const float* b2   = (const float*)d_in[7];
    float* out = (float*)d_out;

    const int E1 = in_sizes[3];          // e1_to_e2 length
    const int E2 = in_sizes[0] / D;      // t_e2 rows

    char* ws = (char*)d_ws;
    int*  winner = (int*)ws;
    bf16* w1bf = (bf16*)(ws + (size_t)E2 * sizeof(int));
    bf16* w2bf = w1bf + 3 * D * D;

    // 1) last-write-wins resolution
    k_init_winner<<<(E2 + 255) / 256, 256, 0, stream>>>(winner, E2);
    k_winner<<<(E1 + 255) / 256, 256, 0, stream>>>(e1m, winner, E1);

    // 2) weights f32 -> bf16 in workspace
    int ncvt = 3 * D * D + D * D;
    k_cvt_w<<<(ncvt + 255) / 256, 256, 0, stream>>>(W1, W2, w1bf, w2bf, 3 * D * D, D * D);

    // 3) bulk copy t_e2 -> t_new
    long n4 = (long)E2 * D / 4;
    k_copy4<<<4096, 256, 0, stream>>>((const float4*)t_e2, (float4*)out, n4);

    // 4) fused gather + MLP(WMMA bf16) + scatter; ~136 KB LDS -> 2 blocks/WGP
    int numTiles = (E1 + 31) / 32;
    size_t ldsBytes =
        (size_t)(384 * WS_STRIDE + 32 * XS_STRIDE + 32 * HS_STRIDE) * sizeof(bf16)
        + 256 * sizeof(float) + 64 * sizeof(int);
    int grid = numTiles < 3072 ? numTiles : 3072;
    k_edge_mlp<<<grid, 256, ldsBytes, stream>>>(t_e2, h, ei, e1m, b1, b2,
                                                w1bf, w2bf, winner, out, E1, numTiles);
}
